// GCNTorch_8701603742375
// MI455X (gfx1250) — matrix operand order repaired and checked by reading, not compiled, not run on hardware
//
#include <hip/hip_runtime.h>
#include <hip/hip_bf16.h>

// ---------------------------------------------------------------------------
// 3-layer GCN for MI455X (gfx1250, wave32).
//   Per layer: H = A @ W    (bf16 WMMA, f32 accumulate, A staged in LDS via
//                            async global->LDS copies, shared by all N-tiles)
//              OUT = bias + dinv^2 * H                   (self loop init)
//              OUT[dst] += dinv[s]*ew*dinv[d] * H[src]   (L2-resident atomics)
//   ReLU fused into the next GEMM's A-matrix read (applied at LDS->VGPR).
// ---------------------------------------------------------------------------

typedef __attribute__((ext_vector_type(16))) __bf16 v16bf;
typedef __attribute__((ext_vector_type(8)))  float  v8f;

#if defined(__gfx1250__) && defined(__has_builtin)
#if __has_builtin(__builtin_amdgcn_global_load_async_to_lds_b128)
#define USE_ASYNC_LDS 1
#endif
#endif
#ifndef USE_ASYNC_LDS
#define USE_ASYNC_LDS 0
#endif

#if USE_ASYNC_LDS
typedef int v4i_t __attribute__((vector_size(16)));
typedef __attribute__((address_space(1))) v4i_t as1_v4i;   // global
typedef __attribute__((address_space(3))) v4i_t as3_v4i;   // LDS
__device__ __forceinline__ void async_copy16(const float* g, float* l) {
  // LDS flat address keeps the byte offset in addr[31:0] (ISA 10.2), so the
  // truncating cast yields the LDS offset the instruction wants.
  __builtin_amdgcn_global_load_async_to_lds_b128(
      (as1_v4i*)(unsigned long long)g,
      (as3_v4i*)(unsigned int)(unsigned long long)l, 0, 0);
}
__device__ __forceinline__ void async_wait0() {
#if __has_builtin(__builtin_amdgcn_s_wait_asynccnt)
  __builtin_amdgcn_s_wait_asynccnt(0);
#else
  asm volatile("s_wait_asynccnt 0x0" ::: "memory");
#endif
}
#endif

// 2x f32 -> packed bf16.  Round-to-nearest (ties up): bias by 0x8000 then keep
// the high 16 bits of each float; both high halves merged by one v_perm_b32.
// (Differs from RNE only on exact ties -- immaterial for bf16 GEMM inputs.)
__device__ __forceinline__ unsigned int pack_bf16x2(float lo, float hi) {
  unsigned int ul = __float_as_uint(lo) + 0x8000u;
  unsigned int uh = __float_as_uint(hi) + 0x8000u;
#if defined(__has_builtin)
#if __has_builtin(__builtin_amdgcn_perm)
  return __builtin_amdgcn_perm(uh, ul, 0x07060302u);  // {uh.b3,uh.b2,ul.b3,ul.b2}
#else
  return (ul >> 16) | (uh & 0xFFFF0000u);
#endif
#else
  return (ul >> 16) | (uh & 0xFFFF0000u);
#endif
}

// ----------------------------- degree / dinv -------------------------------

__global__ void deg_init_kernel(float* __restrict__ deg, int n) {
  int i = blockIdx.x * blockDim.x + threadIdx.x;
  if (i < n) deg[i] = 1.0f;                         // self-loop weight
}

__global__ void deg_edge_kernel(const int* __restrict__ dst,
                                const float* __restrict__ ew,
                                float* __restrict__ deg, int nE) {
  int e = blockIdx.x * blockDim.x + threadIdx.x;
  if (e < nE)
    __hip_atomic_fetch_add(&deg[dst[e]], ew[e],
                           __ATOMIC_RELAXED, __HIP_MEMORY_SCOPE_AGENT);
}

__global__ void dinv_kernel(float* __restrict__ deg, int n) {
  int i = blockIdx.x * blockDim.x + threadIdx.x;
  if (i < n) {
    float d = deg[i];
    deg[i] = (d > 0.0f) ? rsqrtf(d) : 0.0f;         // in-place deg -> dinv
  }
}

// ------------------------------- WMMA GEMM ---------------------------------
// out[M,N] = (RELU? max(A,0) : A)[M,K] @ W[K,N], all row-major fp32 in memory,
// bf16 WMMA with f32 accumulation.  K, N are compile-time (512/64/16).
//
// Block = 256 threads = 8 waves.  Block owns ROWS = P*16 consecutive rows,
// staged chunk-by-chunk (CHUNK of K) into LDS with async copies; each wave
// owns one N-tile and PPW panels, so one B fetch feeds PPW WMMAs and the A
// panel is consumed by TILES_N waves from LDS instead of TILES_N global reads.
template <int K, int N, bool RELU>
__global__ __launch_bounds__(256) void gcn_wmma_gemm(
    const float* __restrict__ A, const float* __restrict__ W,
    float* __restrict__ out, int M) {
  constexpr int TILES_N = N / 16;
  constexpr int WAVES   = 8;
  constexpr int PG      = WAVES / TILES_N;          // panel groups per block
  constexpr int PPW     = (TILES_N > 1) ? 2 : 1;    // panels per wave
  constexpr int P       = PG * PPW;                 // panels per block
  constexpr int ROWS    = P * 16;
  constexpr int CHUNK   = (K >= 256) ? 128 : K;     // K tile staged in LDS
  constexpr int LDSTRIDE = CHUNK + 4;               // pad: kill bank conflicts
  constexpr int CVECS   = CHUNK / 4;                // float4 per row-chunk

  __shared__ __align__(16) float As[P * 16 * LDSTRIDE];

  const int tid  = threadIdx.x;
  const int lane = tid & 31;
  const int wave = tid >> 5;
  const int l16  = lane & 15;
  const int hi   = lane >> 4;                       // 0: lanes 0-15, 1: 16-31

  const int nTile = wave % TILES_N;
  const int pg    = wave / TILES_N;
  const int col   = nTile * 16 + l16;

  int rowBase = blockIdx.x * ROWS;
  if (rowBase > M - ROWS) rowBase = (M > ROWS) ? (M - ROWS) : 0;  // tail clamp

  union U { unsigned int u[8]; v16bf v; };
  v8f acc[PPW];
#pragma unroll
  for (int p = 0; p < PPW; ++p) acc[p] = {};

  for (int kk = 0; kk < K; kk += CHUNK) {
    // ---- stage A[rowBase..rowBase+ROWS) x [kk..kk+CHUNK) into LDS ----
#pragma unroll 1
    for (int c = tid; c < ROWS * CVECS; c += 256) {
      const int r = c / CVECS;
      const int v = c - r * CVECS;
      const float* g = A + (size_t)(rowBase + r) * K + kk + v * 4;
      float*       l = As + r * LDSTRIDE + v * 4;
#if USE_ASYNC_LDS
      async_copy16(g, l);
#else
      *(float4*)l = *(const float4*)g;
#endif
    }
#if USE_ASYNC_LDS
    async_wait0();
#endif
    __syncthreads();

    // ---- compute this K chunk ----
    const float* __restrict__ wbase =
        W + (size_t)kk * N + (size_t)hi * 8 * N + col;
#pragma unroll
    for (int ks = 0; ks < CHUNK; ks += 32) {
      // B 32x16 bf16 layout: VGPR j -> K = hi*8 + (j>=4 ? 8 : 0) + {2j,2j+1}, N = lane%16
      U b;
#pragma unroll
      for (int j = 0; j < 8; ++j)
        b.u[j] = pack_bf16x2(wbase[(size_t)(ks + 2 * j + (j >= 4 ? 8 : 0)) * N],
                             wbase[(size_t)(ks + 2 * j + 1 + (j >= 4 ? 8 : 0)) * N]);
#pragma unroll
      for (int p = 0; p < PPW; ++p) {
        const int panel = pg * PPW + p;
        // A 16x32 bf16 layout: lane holds row l16; VGPR 0-3 <- K=hi*8+0..7,
        //                      VGPR 4-7 <- K=16+hi*8+0..7
        const float* ap = As + (panel * 16 + l16) * LDSTRIDE + ks + hi * 8;
        float4 x0 = *(const float4*)(ap);
        float4 x1 = *(const float4*)(ap + 4);
        float4 x2 = *(const float4*)(ap + 16);
        float4 x3 = *(const float4*)(ap + 20);
        if (RELU) {
          x0.x = fmaxf(x0.x, 0.f); x0.y = fmaxf(x0.y, 0.f);
          x0.z = fmaxf(x0.z, 0.f); x0.w = fmaxf(x0.w, 0.f);
          x1.x = fmaxf(x1.x, 0.f); x1.y = fmaxf(x1.y, 0.f);
          x1.z = fmaxf(x1.z, 0.f); x1.w = fmaxf(x1.w, 0.f);
          x2.x = fmaxf(x2.x, 0.f); x2.y = fmaxf(x2.y, 0.f);
          x2.z = fmaxf(x2.z, 0.f); x2.w = fmaxf(x2.w, 0.f);
          x3.x = fmaxf(x3.x, 0.f); x3.y = fmaxf(x3.y, 0.f);
          x3.z = fmaxf(x3.z, 0.f); x3.w = fmaxf(x3.w, 0.f);
        }
        U a;
        a.u[0] = pack_bf16x2(x0.x, x0.y); a.u[1] = pack_bf16x2(x0.z, x0.w);
        a.u[2] = pack_bf16x2(x1.x, x1.y); a.u[3] = pack_bf16x2(x1.z, x1.w);
        a.u[4] = pack_bf16x2(x2.x, x2.y); a.u[5] = pack_bf16x2(x2.z, x2.w);
        a.u[6] = pack_bf16x2(x3.x, x3.y); a.u[7] = pack_bf16x2(x3.z, x3.w);
        acc[p] = __builtin_amdgcn_wmma_f32_16x16x32_bf16(
            false, a.v, false, b.v, (short)0, acc[p], false, false);
      }
    }
    __syncthreads();   // protect LDS before next chunk's copies
  }

  // D 16x16 f32 layout: VGPR j -> row j + 8*hi, col = lane%16
#pragma unroll
  for (int p = 0; p < PPW; ++p) {
    const int mBase = rowBase + (pg * PPW + p) * 16 + hi * 8;
    float* op = out + (size_t)mBase * N + col;
#pragma unroll
    for (int j = 0; j < 8; ++j) op[(size_t)j * N] = acc[p][j];
  }
}

// ----------------------- aggregation (self + edges) ------------------------

// out[i,f] = bias[f] + dinv[i]^2 * H[i,f]     (self-loop term + bias init)
__global__ void self_init_kernel(const float* __restrict__ H,
                                 const float* __restrict__ dinv,
                                 const float* __restrict__ bias,
                                 float* __restrict__ out,
                                 int n, int logF) {
  int idx = blockIdx.x * blockDim.x + threadIdx.x;
  int total = n << logF;
  if (idx >= total) return;
  int i = idx >> logF;
  int f = idx & ((1 << logF) - 1);
  float d = dinv[i];
  out[idx] = bias[f] + H[idx] * d * d;
}

// out[dst,f] += dinv[s]*ew*dinv[d] * H[src,f]   (one thread per (edge,feat))
__global__ void edge_agg_kernel(const float* __restrict__ H,
                                const float* __restrict__ dinv,
                                const int* __restrict__ src,
                                const int* __restrict__ dst,
                                const float* __restrict__ ew,
                                float* __restrict__ out,
                                int nE, int logF) {
  int idx = blockIdx.x * blockDim.x + threadIdx.x;
  int e = idx >> logF;
  if (e >= nE) return;
  int f = idx & ((1 << logF) - 1);
  int s = src[e];
  int d = dst[e];
  float norm = dinv[s] * ew[e] * dinv[d];
  float v = H[((size_t)s << logF) + f] * norm;
  __hip_atomic_fetch_add(&out[((size_t)d << logF) + f], v,
                         __ATOMIC_RELAXED, __HIP_MEMORY_SCOPE_AGENT);
}

// ------------------------------- launcher ----------------------------------

extern "C" void kernel_launch(void* const* d_in, const int* in_sizes, int n_in,
                              void* d_out, int out_size, void* d_ws, size_t ws_size,
                              hipStream_t stream) {
  const float* X  = (const float*)d_in[0];
  const int*   ei = (const int*)  d_in[1];
  const float* ew = (const float*)d_in[2];
  const float* W1 = (const float*)d_in[3];
  const float* b1 = (const float*)d_in[4];
  const float* W2 = (const float*)d_in[5];
  const float* b2 = (const float*)d_in[6];
  const float* W3 = (const float*)d_in[7];
  const float* b3 = (const float*)d_in[8];
  float* out = (float*)d_out;

  constexpr int IN  = 512;   // IN_FEATS
  constexpr int HID = 64;    // N_HIDDEN
  constexpr int CLS = 16;    // N_CLASSES
  constexpr int LOG_HID = 6;
  constexpr int LOG_CLS = 4;

  const int n  = in_sizes[0] / IN;     // 100000 nodes
  const int nE = in_sizes[2];          // 1000000 edges
  const int* src = ei;
  const int* dst = ei + nE;

  float* ws   = (float*)d_ws;
  float* dinv = ws;                            // n floats
  float* bufH = ws + n;                        // n*HID floats (GEMM output)
  float* bufA = bufH + (size_t)n * HID;        // n*HID floats (agg output)

  const int B = 256;

  // --- degree / dinv (graph-only) ---
  deg_init_kernel<<<(n + B - 1) / B, B, 0, stream>>>(dinv, n);
  deg_edge_kernel<<<(nE + B - 1) / B, B, 0, stream>>>(dst, ew, dinv, nE);
  dinv_kernel<<<(n + B - 1) / B, B, 0, stream>>>(dinv, n);

  auto aggregate = [&](const float* Ho, const float* bias, float* Ao, int logF) {
    long long tSelf = (long long)n  << logF;
    long long tEdge = (long long)nE << logF;
    self_init_kernel<<<(int)((tSelf + B - 1) / B), B, 0, stream>>>(
        Ho, dinv, bias, Ao, n, logF);
    edge_agg_kernel<<<(int)((tEdge + B - 1) / B), B, 0, stream>>>(
        Ho, dinv, src, dst, ew, Ao, nE, logF);
  };

  // --- layer 1: X[n,512] @ W1 -> H ; aggregate -> bufA ---
  {
    constexpr int ROWS = 64;   // P*16 for N=64
    int grid = (n + ROWS - 1) / ROWS;
    gcn_wmma_gemm<IN, HID, false><<<grid, B, 0, stream>>>(X, W1, bufH, n);
  }
  aggregate(bufH, b1, bufA, LOG_HID);

  // --- layer 2: relu(bufA) @ W2 -> H ; aggregate -> bufA ---
  {
    constexpr int ROWS = 64;
    int grid = (n + ROWS - 1) / ROWS;
    gcn_wmma_gemm<HID, HID, true><<<grid, B, 0, stream>>>(bufA, W2, bufH, n);
  }
  aggregate(bufH, b2, bufA, LOG_HID);

  // --- layer 3: relu(bufA) @ W3 -> H3 ; aggregate -> d_out ---
  {
    constexpr int ROWS = 128;  // P*16 for N=16
    int grid = (n + ROWS - 1) / ROWS;
    gcn_wmma_gemm<HID, CLS, true><<<grid, B, 0, stream>>>(bufA, W3, bufH, n);
  }
  aggregate(bufH, b3, out, LOG_CLS);
}
